// LlamaAttention_2800318677157
// MI455X (gfx1250) — compile-verified, matrix operand order checked by reading, not run on hardware
//
#include <hip/hip_runtime.h>
#include <cstdint>
#include <cstddef>

// ---------------- constants (match reference) ----------------
#define T_LEN   2048
#define D_DIM   4096
#define NQ      32
#define NKV     8
#define HD      128
#define S_CACHE 4096
#define LN_THETA 13.122363377404328f   // ln(500000)
#define QSCALE   0.08838834764831845f  // 128^-0.5
#define NEG_INF_F (-1.0e30f)

typedef __bf16 bf16_t;
typedef __attribute__((ext_vector_type(16))) __bf16 v16bf;
typedef __attribute__((ext_vector_type(8)))  __bf16 v8bf;
typedef __attribute__((ext_vector_type(8)))  float  v8f;
typedef __attribute__((ext_vector_type(4)))  int    v4i;

#if defined(__has_builtin)
#  if __has_builtin(__builtin_amdgcn_global_load_async_to_lds_b128)
#    define HAVE_ASYNC_LDS 1
#  endif
#endif
#ifndef HAVE_ASYNC_LDS
#  define HAVE_ASYNC_LDS 0
#endif

typedef __attribute__((address_space(1))) v4i* gv4i_p;  // global int4*
typedef __attribute__((address_space(3))) v4i* sv4i_p;  // LDS int4*

__device__ __forceinline__ bf16_t f2bf(float f) {
    unsigned u = __builtin_bit_cast(unsigned, f);
    unsigned r = (u + 0x7FFFu + ((u >> 16) & 1u)) >> 16;
    return __builtin_bit_cast(bf16_t, (unsigned short)r);
}

// A/B fragment (16x32 bf16), per ISA 16-bit A-matrix layout:
// lane<16 : row = lane,    K = {0..7, 16..23}
// lane>=16: row = lane-16, K = {8..15, 24..31}
// caller passes p = base + row*ld + kbase + b0 (b0 = (lane>>4)*8); 16B aligned.
__device__ __forceinline__ v16bf load_frag(const bf16_t* p) {
    v8bf lo = *(const v8bf*)(p);
    v8bf hi = *(const v8bf*)(p + 16);
    return __builtin_shufflevector(lo, hi, 0,1,2,3,4,5,6,7,8,9,10,11,12,13,14,15);
}

__device__ __forceinline__ v8f wmma_bf16(v16bf a, v16bf b, v8f c) {
    return __builtin_amdgcn_wmma_f32_16x16x32_bf16(false, a, false, b, (short)0, c, false, false);
}

// 16-byte global -> LDS copy (async on CDNA5; ASYNCcnt-tracked)
__device__ __forceinline__ void async_copy16(const bf16_t* g, bf16_t* l) {
#if HAVE_ASYNC_LDS
    __builtin_amdgcn_global_load_async_to_lds_b128(
        (gv4i_p)const_cast<bf16_t*>(g), (sv4i_p)l, 0, 0);
#else
    *(v8bf*)l = *(const v8bf*)g;
#endif
}

__device__ __forceinline__ void wait_async0() {
#if HAVE_ASYNC_LDS
    asm volatile("s_wait_asynccnt 0x0" ::: "memory");
#endif
}

// ---------------- shared GEMM mainloop ----------------
// C_tile(128x128) += A[rbase..rbase+127, :Kdim] * Bt[0..127, :Kdim]^T
// B^T slab (128 rows x 32 k, 8KB) staged to LDS per block, double-buffered,
// via async global->LDS; each of 8 waves computes 16 rows x 128 cols.
// lds: 2*128*32 bf16 (16 KB). abase pre-offset to this wave's 16 rows.
__device__ __forceinline__ void gemm_mainloop_128(
    const bf16_t* __restrict__ abase, int lda,
    const bf16_t* __restrict__ bt, int ldb, int Kdim,
    bf16_t* lds, v8f acc[8]) {
    const int tid = threadIdx.x;
    const int lane = tid & 31;
    const int lane15 = lane & 15;
    const int b0 = (lane >> 4) * 8;
    const int niter = Kdim / 32;

    // stage one 128x32 slab: 512 chunks of 16B; 2 chunks per thread
    auto stage = [&](int buf, int kk) {
        bf16_t* dst = lds + buf * (128 * 32);
#pragma unroll
        for (int j = 0; j < 2; ++j) {
            int c = tid + j * 256;
            int row = c >> 2, part = (c & 3) * 8;
            async_copy16(bt + (size_t)row * ldb + kk + part, dst + row * 32 + part);
        }
    };

    stage(0, 0);
    const bf16_t* arow = abase + (size_t)lane15 * lda;
    for (int i = 0; i < niter; ++i) {
        wait_async0();        // my part of stage i landed in LDS
        __syncthreads();      // everyone's stage i landed; stage i-1 consumers done
        if (i + 1 < niter) stage((i + 1) & 1, (i + 1) * 32);
        const bf16_t* cur = lds + (i & 1) * (128 * 32);
        const int kk = i * 32;
        v16bf a = load_frag(arow + kk + b0);
#pragma unroll
        for (int nt = 0; nt < 8; ++nt) {
            v16bf b = load_frag(cur + (nt * 16 + lane15) * 32 + b0);
            acc[nt] = wmma_bf16(a, b, acc[nt]);
        }
    }
}

// ---------------- prep kernels ----------------
__global__ void cvt_f32_bf16(const float* __restrict__ in, bf16_t* __restrict__ out, int n) {
    int stride = gridDim.x * blockDim.x;
    for (int i = blockIdx.x * blockDim.x + threadIdx.x; i < n; i += stride)
        out[i] = f2bf(in[i]);
}

// per-head transpose: in[head][r][c] -> out[head][c][r] (fp32 -> bf16)
__global__ void tconv_kernel(const float* __restrict__ in, bf16_t* __restrict__ out,
                             int heads, int rows, int cols) {
    int hc = rows * cols;
    int total = heads * hc;
    int stride = gridDim.x * blockDim.x;
    for (int i = blockIdx.x * blockDim.x + threadIdx.x; i < total; i += stride) {
        int h = i / hc; int rem = i - h * hc;
        int r = rem / cols; int c = rem - r * cols;
        out[(size_t)h * hc + (size_t)c * rows + r] = f2bf(in[i]);
    }
}

// ---------------- QKV projection + RoPE + cache scatter ----------------
// grid: (T/128, NQ + 2*NKV); block 256 (8 waves); wave computes 16 rows x 128 cols
__global__ __launch_bounds__(256)
void qkv_rope_kernel(const bf16_t* __restrict__ xbf,
                     const bf16_t* __restrict__ wqt,   // [NQ][HD][D]
                     const bf16_t* __restrict__ wkt,   // [NKV][HD][D]
                     const bf16_t* __restrict__ wvt,   // [NKV][HD][D]
                     const int*    __restrict__ positions,   // [T]
                     const int*    __restrict__ wridx,       // [T]
                     bf16_t* __restrict__ qbf,   // [NQ][T][HD]
                     bf16_t* __restrict__ kbf,   // [NKV][T][HD]
                     bf16_t* __restrict__ vtbf,  // [NKV][HD][T]
                     float*  __restrict__ kcache,  // [NKV][S_CACHE][HD]
                     float*  __restrict__ vcache) {
    __shared__ __align__(16) bf16_t ldsbuf[2 * 128 * 32];

    const int wave = threadIdx.x >> 5;
    const int lane = threadIdx.x & 31;
    const int lane15 = lane & 15;
    const int rowm = (lane >> 4) * 8;

    const int hy = blockIdx.y;
    const int rbase = blockIdx.x * 128 + wave * 16;

    const bf16_t* wt;
    if (hy < NQ)            wt = wqt + (size_t)hy * HD * D_DIM;
    else if (hy < NQ + NKV) wt = wkt + (size_t)(hy - NQ) * HD * D_DIM;
    else                    wt = wvt + (size_t)(hy - NQ - NKV) * HD * D_DIM;

    v8f acc[8];
#pragma unroll
    for (int i = 0; i < 8; ++i) acc[i] = (v8f){0,0,0,0,0,0,0,0};

    gemm_mainloop_128(xbf + (size_t)rbase * D_DIM, D_DIM, wt, D_DIM, D_DIM, ldsbuf, acc);

    if (hy < NQ + NKV) {
        // RoPE (split-half): pair (h, h+64) -> (acc[nt], acc[nt+4]), same lane/element
        const bool isq = (hy < NQ);
        const float scale = isq ? QSCALE : 1.0f;
#pragma unroll
        for (int r = 0; r < 8; ++r) {
            int t = rbase + rowm + r;
            float pos = (float)positions[t];
#pragma unroll
            for (int nt = 0; nt < 4; ++nt) {
                int h1 = nt * 16 + lane15;
                float freq = __expf(-(float)h1 * (LN_THETA / 64.0f));
                float ang = pos * freq;
                float cs = __cosf(ang), sn = __sinf(ang);
                float x1 = acc[nt][r], x2 = acc[nt + 4][r];
                acc[nt][r]     = (x1 * cs - x2 * sn) * scale;
                acc[nt + 4][r] = (x2 * cs + x1 * sn) * scale;
            }
        }
        if (isq) {
            bf16_t* dst = qbf + (size_t)hy * T_LEN * HD;
#pragma unroll
            for (int nt = 0; nt < 8; ++nt)
#pragma unroll
                for (int r = 0; r < 8; ++r) {
                    int t = rbase + rowm + r, h = nt * 16 + lane15;
                    dst[(size_t)t * HD + h] = f2bf(acc[nt][r]);
                }
        } else {
            int g = hy - NQ;
#pragma unroll
            for (int nt = 0; nt < 8; ++nt)
#pragma unroll
                for (int r = 0; r < 8; ++r) {
                    int t = rbase + rowm + r, h = nt * 16 + lane15;
                    float v = acc[nt][r];
                    kbf[((size_t)g * T_LEN + t) * HD + h] = f2bf(v);
                    int slot = wridx[t];
                    kcache[((size_t)g * S_CACHE + slot) * HD + h] = v;
                }
        }
    } else {
        int g = hy - NQ - NKV;
#pragma unroll
        for (int nt = 0; nt < 8; ++nt)
#pragma unroll
            for (int r = 0; r < 8; ++r) {
                int t = rbase + rowm + r, h = nt * 16 + lane15;
                float v = acc[nt][r];
                int slot = wridx[t];
                vcache[((size_t)g * S_CACHE + slot) * HD + h] = v;
                vtbf[((size_t)g * HD + h) * T_LEN + t] = f2bf(v);
            }
    }
}

// ---------------- causal flash attention (online softmax) ----------------
// grid: (T/128, NQ); block 256; wave handles 16 query rows x HD cols
__global__ __launch_bounds__(256)
void attn_kernel(const bf16_t* __restrict__ qbf,   // [NQ][T][HD]
                 const bf16_t* __restrict__ kbf,   // [NKV][T][HD]
                 const bf16_t* __restrict__ vtbf,  // [NKV][HD][T]
                 bf16_t* __restrict__ attnbf) {    // [T][NQ*HD]
    __shared__ __align__(16) bf16_t pb[8 * 16 * 32];

    const int wave = threadIdx.x >> 5;
    const int lane = threadIdx.x & 31;
    const int lane15 = lane & 15;
    const int b0 = (lane >> 4) * 8;
    const int rowm = (lane >> 4) * 8;

    const int n = blockIdx.y;
    const int g = n >> 2;                 // GQA: 32 q heads -> 8 kv heads
    const int qbase = blockIdx.x * 128 + wave * 16;

    // preload Q fragments (16 x 128, four K=32 slabs)
    const bf16_t* qrow = qbf + ((size_t)n * T_LEN + qbase + lane15) * HD;
    v16bf qa[4];
#pragma unroll
    for (int kk = 0; kk < 4; ++kk) qa[kk] = load_frag(qrow + kk * 32 + b0);

    v8f oacc[8];
#pragma unroll
    for (int i = 0; i < 8; ++i) oacc[i] = (v8f){0,0,0,0,0,0,0,0};
    float rmax[8], rsum[8];
#pragma unroll
    for (int r = 0; r < 8; ++r) { rmax[r] = -3.0e38f; rsum[r] = 0.0f; }

    bf16_t* myp = pb + wave * (16 * 32);
    const bf16_t* vbase = vtbf + (size_t)g * HD * T_LEN;
    const int nchunks = (qbase + 16 + 31) >> 5;

    for (int c = 0; c < nchunks; ++c) {
        const int sbase = c * 32;
        v8f s0 = (v8f){0,0,0,0,0,0,0,0};
        v8f s1 = (v8f){0,0,0,0,0,0,0,0};
        const bf16_t* k0 = kbf + ((size_t)g * T_LEN + sbase + lane15) * HD;
        const bf16_t* k1 = k0 + 16 * HD;
#pragma unroll
        for (int kk = 0; kk < 4; ++kk) {
            int off = kk * 32 + b0;
            s0 = wmma_bf16(qa[kk], load_frag(k0 + off), s0);
            s1 = wmma_bf16(qa[kk], load_frag(k1 + off), s1);
        }
        // causal mask + online softmax (rows spread over 16-lane half-groups)
        float pv0[8], pv1[8], alpha[8];
#pragma unroll
        for (int r = 0; r < 8; ++r) {
            int t = qbase + rowm + r;
            int sA = sbase + lane15, sB = sA + 16;
            float v0 = (sA <= t) ? s0[r] : NEG_INF_F;
            float v1 = (sB <= t) ? s1[r] : NEG_INF_F;
            float mloc = fmaxf(v0, v1);
#pragma unroll
            for (int m = 1; m < 16; m <<= 1) mloc = fmaxf(mloc, __shfl_xor(mloc, m, 32));
            float mnew = fmaxf(rmax[r], mloc);
            float al = __expf(rmax[r] - mnew);
            float p0 = __expf(v0 - mnew);
            float p1 = __expf(v1 - mnew);
            float ps = p0 + p1;
#pragma unroll
            for (int m = 1; m < 16; m <<= 1) ps += __shfl_xor(ps, m, 32);
            rsum[r] = rsum[r] * al + ps;
            rmax[r] = mnew;
            alpha[r] = al; pv0[r] = p0; pv1[r] = p1;
        }
#pragma unroll
        for (int nt = 0; nt < 8; ++nt)
#pragma unroll
            for (int r = 0; r < 8; ++r) oacc[nt][r] *= alpha[r];

        // re-layout P (C-frag) -> A-frag via LDS (intra-wave transpose)
#pragma unroll
        for (int r = 0; r < 8; ++r) {
            int m = rowm + r;
            myp[m * 32 + lane15]      = f2bf(pv0[r]);
            myp[m * 32 + 16 + lane15] = f2bf(pv1[r]);
        }
        asm volatile("s_wait_dscnt 0" ::: "memory");
        v16bf pa = load_frag(myp + lane15 * 32 + b0);

#pragma unroll
        for (int nt = 0; nt < 8; ++nt) {
            v16bf bv = load_frag(vbase + (size_t)(nt * 16 + lane15) * T_LEN + sbase + b0);
            oacc[nt] = wmma_bf16(pa, bv, oacc[nt]);
        }
    }

    // normalize and emit attn (bf16, [T][NQ*HD])
#pragma unroll
    for (int nt = 0; nt < 8; ++nt)
#pragma unroll
        for (int r = 0; r < 8; ++r) {
            int t = qbase + rowm + r, h = nt * 16 + lane15;
            float inv = 1.0f / rsum[r];
            attnbf[(size_t)t * (NQ * HD) + n * HD + h] = f2bf(oacc[nt][r] * inv);
        }
}

// ---------------- output projection ----------------
// grid: (T/128, D/128); o[t,d] = attn[t,:] . woT[d,:]
__global__ __launch_bounds__(256)
void oproj_kernel(const bf16_t* __restrict__ attnbf,  // [T][4096]
                  const bf16_t* __restrict__ wot,     // [D][4096]
                  float* __restrict__ out) {          // [T][D]
    __shared__ __align__(16) bf16_t ldsbuf[2 * 128 * 32];

    const int wave = threadIdx.x >> 5;
    const int lane = threadIdx.x & 31;
    const int lane15 = lane & 15;
    const int rowm = (lane >> 4) * 8;
    const int rbase = blockIdx.x * 128 + wave * 16;
    const int dbase = blockIdx.y * 128;
    const int KD = NQ * HD;  // 4096

    v8f acc[8];
#pragma unroll
    for (int i = 0; i < 8; ++i) acc[i] = (v8f){0,0,0,0,0,0,0,0};

    gemm_mainloop_128(attnbf + (size_t)rbase * KD, KD,
                      wot + (size_t)dbase * KD, KD, KD, ldsbuf, acc);

#pragma unroll
    for (int nt = 0; nt < 8; ++nt)
#pragma unroll
        for (int r = 0; r < 8; ++r) {
            int t = rbase + rowm + r, d = dbase + nt * 16 + lane15;
            out[(size_t)t * D_DIM + d] = acc[nt][r];
        }
}

// ---------------- host-side launch ----------------
extern "C" void kernel_launch(void* const* d_in, const int* in_sizes, int n_in,
                              void* d_out, int out_size, void* d_ws, size_t ws_size,
                              hipStream_t stream) {
    const float* x   = (const float*)d_in[0];
    const float* wq  = (const float*)d_in[1];
    const float* wk  = (const float*)d_in[2];
    const float* wv  = (const float*)d_in[3];
    const float* wo  = (const float*)d_in[4];
    const float* kci = (const float*)d_in[5];
    const float* vci = (const float*)d_in[6];
    const int* positions = (const int*)d_in[7];
    const int* wridx     = (const int*)d_in[8];

    float* o_out = (float*)d_out;                                   // [T][D]
    float* kc_out = o_out + (size_t)T_LEN * D_DIM;                  // [NKV][S_CACHE][HD]
    float* vc_out = kc_out + (size_t)NKV * S_CACHE * HD;

    // workspace layout (bf16)
    char* w = (char*)d_ws;
    size_t off = 0;
    bf16_t* xbf  = (bf16_t*)(w + off); off += (size_t)T_LEN * D_DIM * 2;          // 16 MB
    bf16_t* wqt  = (bf16_t*)(w + off); off += (size_t)NQ  * HD * D_DIM * 2;       // 32 MB
    bf16_t* wkt  = (bf16_t*)(w + off); off += (size_t)NKV * HD * D_DIM * 2;       //  8 MB
    bf16_t* wvt  = (bf16_t*)(w + off); off += (size_t)NKV * HD * D_DIM * 2;       //  8 MB
    bf16_t* wot  = (bf16_t*)(w + off); off += (size_t)D_DIM * NQ * HD * 2;        // 32 MB
    bf16_t* qbf  = (bf16_t*)(w + off); off += (size_t)NQ  * T_LEN * HD * 2;       // 16 MB
    bf16_t* kbf  = (bf16_t*)(w + off); off += (size_t)NKV * T_LEN * HD * 2;       //  4 MB
    bf16_t* vtbf = (bf16_t*)(w + off); off += (size_t)NKV * HD * T_LEN * 2;       //  4 MB
    bf16_t* attnbf = (bf16_t*)(w + off);                                          // 16 MB

    // caches: copy inputs, then scatter written rows
    (void)hipMemcpyAsync(kc_out, kci, (size_t)NKV * S_CACHE * HD * sizeof(float),
                         hipMemcpyDeviceToDevice, stream);
    (void)hipMemcpyAsync(vc_out, vci, (size_t)NKV * S_CACHE * HD * sizeof(float),
                         hipMemcpyDeviceToDevice, stream);

    // prep: conversions / transposes
    cvt_f32_bf16<<<4096, 256, 0, stream>>>(x, xbf, T_LEN * D_DIM);
    tconv_kernel<<<8192, 256, 0, stream>>>(wq, wqt, NQ,  D_DIM, HD);
    tconv_kernel<<<4096, 256, 0, stream>>>(wk, wkt, NKV, D_DIM, HD);
    tconv_kernel<<<4096, 256, 0, stream>>>(wv, wvt, NKV, D_DIM, HD);
    tconv_kernel<<<8192, 256, 0, stream>>>(wo, wot, 1, NQ * HD, D_DIM);

    // QKV + RoPE + cache scatter
    qkv_rope_kernel<<<dim3(T_LEN / 128, NQ + 2 * NKV), 256, 0, stream>>>(
        xbf, wqt, wkt, wvt, positions, wridx, qbf, kbf, vtbf, kc_out, vc_out);

    // causal flash attention
    attn_kernel<<<dim3(T_LEN / 128, NQ), 256, 0, stream>>>(qbf, kbf, vtbf, attnbf);

    // output projection
    oproj_kernel<<<dim3(T_LEN / 128, D_DIM / 128), 256, 0, stream>>>(attnbf, wot, o_out);
}
